// SUModule_8761733284508
// MI455X (gfx1250) — matrix operand order repaired and checked by reading, not compiled, not run on hardware
//
#include <hip/hip_runtime.h>

// ---------------------------------------------------------------------------
// PointNet++-style set-upconv for MI455X (gfx1250, wave32, WMMA).
//
// Pipeline (all on `stream`):
//   K1  knn            -> idx (B,N1,16)                       [ws]
//   K2  layer0 GEMM    -> per-block partial (sum,sumsq) of y0 [ws, no atomics]
//   K2b reduce+fold BN0 -> scale0/shift0
//   K3  layer0+affine+relu -> layer1 GEMM -> partial stats y1
//   K3b reduce+fold BN1 -> scale1/shift1
//   K4  layer0->layer1 (affine+relu both) -> max over k -> f  [ws, 16 MB]
//   K5  layer2 GEMM on [f; feature1] -> pre-BN y2 to d_out + partial stats
//   K5b reduce+fold BN2 -> scale2/shift2
//   K6  in-place affine+relu on d_out
//
// All matmuls use v_wmma_f32_16x16x32_f16 (f16 in, f32 accumulate).
// ---------------------------------------------------------------------------

#define DEV __device__ __forceinline__

typedef _Float16 h16;
typedef __attribute__((ext_vector_type(8)))  _Float16 v8h;
typedef __attribute__((ext_vector_type(16))) _Float16 v16h;
typedef __attribute__((ext_vector_type(8)))  float    v8f;

constexpr int B    = 8;
constexpr int N1   = 8192;
constexpr int N2   = 2048;
constexpr int KNNS = 16;     // neighbors
constexpr int CIN1 = 67;     // F2 + 3
constexpr int CP1  = 96;     // layer0 K padded to 3x32
constexpr int M0   = 64;
constexpr int M1   = 64;
constexpr int CIN2 = 128;
constexpr int M2   = 128;
constexpr float BN_EPS = 1e-5f;

DEV int lane_id() { return (int)(threadIdx.x & 31); }

DEV v16h cat16(v8h lo, v8h hi) {
  return (v16h)__builtin_shufflevector(lo, hi, 0,1,2,3,4,5,6,7,8,9,10,11,12,13,14,15);
}

// A fragment (16x32 f16): row-major LDS tile, leading dim `ld` halves.
// lane<16 holds row (lane), elements 0..7 = K[off..off+7], 8..15 = K[16+off..],
// off = (lane>=16)*8  (matches CDNA5 16-bit A layout).
DEV v16h load_a(const h16* W, int ld, int row0, int k0) {
  int l = lane_id();
  const h16* p = W + (row0 + (l & 15)) * ld + k0 + ((l >> 4) << 3);
  v8h lo = *(const v8h*)(p);
  v8h hi = *(const v8h*)(p + 16);
  return cat16(lo, hi);
}

// B fragment (32x16 f16): column-major LDS tile (column stride `ld` halves).
// lane holds column (lane&15), elements 0..7 = K[off..off+7], 8..15 = K[16+off..], off = (lane>=16)*8.
DEV v16h load_b(const h16* X, int ld, int k0) {
  int l = lane_id();
  const h16* p = X + (l & 15) * ld + k0 + ((l >> 4) << 3);
  v8h lo = *(const v8h*)(p);
  v8h hi = *(const v8h*)(p + 16);
  return cat16(lo, hi);
}

DEV v8f wmma_f16(v16h a, v16h b, v8f c) {
  return __builtin_amdgcn_wmma_f32_16x16x32_f16(false, a, false, b, (short)0, c,
                                                false, false);
}

// acc += W(NMT*16 x NKT*32) * X(NKT*32 x 16); both tiles use leading dim LD.
template<int NKT, int NMT, int LD>
DEV void mm_tiles(const h16* W, const h16* X, v8f* acc) {
#pragma unroll
  for (int kt = 0; kt < NKT; ++kt) {
    v16h bf = load_b(X, LD, kt * 32);
#pragma unroll
    for (int mt = 0; mt < NMT; ++mt) {
      v16h af = load_a(W, LD, mt * 16, kt * 32);
      acc[mt] = wmma_f16(af, bf, acc[mt]);
    }
  }
}

// Per-wave deterministic (sum, sumsq) of accumulator rows into per-wave LDS
// slots sS[M], sQ[M] (M = channel row). 16-lane shfl tree, fixed order.
template<int NMT>
DEV void wave_stats(const v8f* acc, float* sS, float* sQ) {
  int lane = lane_id(), hi = lane >> 4;
#pragma unroll
  for (int mt = 0; mt < NMT; ++mt) {
#pragma unroll
    for (int r = 0; r < 8; ++r) {
      float v = acc[mt][r];
      float s = v, q = v * v;
#pragma unroll
      for (int o = 1; o < 16; o <<= 1) {
        s += __shfl_xor(s, o, 32);
        q += __shfl_xor(q, o, 32);
      }
      if ((lane & 15) == 0) {
        int M = mt * 16 + hi * 8 + r;
        sS[M] = s;
        sQ[M] = q;
      }
    }
  }
}

// Gather one query's 16-neighbor layer0 input tile into column-major LDS
// (16 columns x CP1 halves).  x = [feature2[:, m]; pos2[:, m]-pos1[:, n]; 0pad]
DEV void gather_x(h16* xt, const float* f2b, const float* p2b,
                  float p1x, float p1y, float p1z, const int* idxn) {
  int l = lane_id();
  int mj = (l < 16) ? idxn[l] : 0;
  for (int j = 0; j < 16; ++j) {
    int m = __shfl(mj, j, 32);
    h16* col = xt + j * CP1;
#pragma unroll
    for (int k = l; k < CP1; k += 32) {
      float v;
      if (k < 64)       v = f2b[k * N2 + m];
      else if (k == 64) v = p2b[0 * N2 + m] - p1x;
      else if (k == 65) v = p2b[1 * N2 + m] - p1y;
      else if (k == 66) v = p2b[2 * N2 + m] - p1z;
      else              v = 0.f;
      col[k] = (h16)v;
    }
  }
}

// --------------------------------------------------------------------- K1 ---
__global__ void __launch_bounds__(256)
k_knn(const float* __restrict__ pos1, const float* __restrict__ pos2,
      int* __restrict__ idx) {
  __shared__ float px[N2], py[N2], pz[N2];
  int b = (int)blockIdx.x >> 5;            // 32 blocks per batch
  int nbase = ((int)blockIdx.x & 31) << 8; // 256 queries per block
  const float* p2b = pos2 + (size_t)b * 3 * N2;
  for (int i = threadIdx.x; i < N2; i += 256) {
    px[i] = p2b[i];
    py[i] = p2b[N2 + i];
    pz[i] = p2b[2 * N2 + i];
  }
  __syncthreads();

  int n = nbase + (int)threadIdx.x;
  float qx = pos1[(b * 3 + 0) * N1 + n];
  float qy = pos1[(b * 3 + 1) * N1 + n];
  float qz = pos1[(b * 3 + 2) * N1 + n];

  float bd[16];
  int   bi[16];
#pragma unroll
  for (int i = 0; i < 16; ++i) { bd[i] = 3.0e38f; bi[i] = 0; }

  for (int m = 0; m < N2; ++m) {
    float dx = qx - px[m], dy = qy - py[m], dz = qz - pz[m];
    float d = fmaf(dx, dx, fmaf(dy, dy, dz * dz));
    if (d < bd[15]) {
      // fully-unrolled select-chain insertion (registers only)
#pragma unroll
      for (int j = 15; j >= 1; --j) {
        bool shift = d < bd[j - 1];
        bool place = (!shift) && (d < bd[j]);
        float nb = shift ? bd[j - 1] : (place ? d : bd[j]);
        int   ni = shift ? bi[j - 1] : (place ? m : bi[j]);
        bd[j] = nb;
        bi[j] = ni;
      }
      if (d < bd[0]) { bd[0] = d; bi[0] = m; }
    }
  }
  int* op = idx + ((size_t)(b * N1 + n)) * KNNS;
#pragma unroll
  for (int s = 0; s < 16; ++s) op[s] = bi[s];
}

// --------------------------------------------------------------------- K2 ---
// y0 = W0 * X, accumulate per-block channel stats only (y0 discarded).
__global__ void __launch_bounds__(256)
k_l0_stats(const float* __restrict__ pos1, const float* __restrict__ pos2,
           const float* __restrict__ feature2, const float* __restrict__ w1_0,
           const int* __restrict__ idx,
           float* __restrict__ psum, float* __restrict__ psq) {
  __shared__ __align__(16) h16 sW[M0 * CP1];        // 12 KB
  __shared__ __align__(16) h16 sX[8 * 16 * CP1];    // 24 KB
  __shared__ float sS[8 * 64], sQ[8 * 64];          // 4 KB
  int tid = (int)threadIdx.x;
  for (int i = tid; i < M0 * CP1; i += 256) {
    int r = i / CP1, k = i - r * CP1;
    sW[i] = (h16)((k < CIN1) ? w1_0[r * CIN1 + k] : 0.f);
  }
  __syncthreads();

  int wave = tid >> 5;
  int tile = (int)blockIdx.x * 8 + wave;            // one query per wave
  int b = tile >> 13, n = tile & (N1 - 1);
  const float* f2b = feature2 + (size_t)b * 64 * N2;
  const float* p2b = pos2 + (size_t)b * 3 * N2;
  float p1x = pos1[(b * 3 + 0) * N1 + n];
  float p1y = pos1[(b * 3 + 1) * N1 + n];
  float p1z = pos1[(b * 3 + 2) * N1 + n];
  h16* xt = sX + wave * (16 * CP1);
  gather_x(xt, f2b, p2b, p1x, p1y, p1z, idx + (size_t)tile * KNNS);
  __syncthreads();

  v8f zero = {0.f, 0.f, 0.f, 0.f, 0.f, 0.f, 0.f, 0.f};
  v8f acc[4] = {zero, zero, zero, zero};
  mm_tiles<3, 4, CP1>(sW, xt, acc);

  wave_stats<4>(acc, sS + wave * 64, sQ + wave * 64);
  __syncthreads();
  if (tid < 64) {
    float s = 0.f, q = 0.f;
#pragma unroll
    for (int w = 0; w < 8; ++w) { s += sS[w * 64 + tid]; q += sQ[w * 64 + tid]; }
    psum[(size_t)blockIdx.x * 64 + tid] = s;
    psq [(size_t)blockIdx.x * 64 + tid] = q;
  }
}

// ------------------------------------------------------------------ K3/K4 ---
// h0 = relu(scale0*y0+shift0); y1 = W1*h0.
// WRITE_F=false: emit per-block stats of y1.
// WRITE_F=true : apply BN1+relu, max over 16 neighbors, write f (B,64,N1).
template<bool WRITE_F>
__global__ void __launch_bounds__(256)
k_mlp1(const float* __restrict__ pos1, const float* __restrict__ pos2,
       const float* __restrict__ feature2, const float* __restrict__ w1_0,
       const float* __restrict__ w1_1, const int* __restrict__ idx,
       const float* __restrict__ scale0, const float* __restrict__ shift0,
       const float* __restrict__ scale1, const float* __restrict__ shift1,
       float* __restrict__ psum, float* __restrict__ psq,
       float* __restrict__ fbuf) {
  __shared__ __align__(16) h16 sW0[M0 * CP1];       // 12 KB
  __shared__ __align__(16) h16 sW1[M1 * M0];        //  8 KB
  __shared__ __align__(16) h16 sX[8 * 16 * CP1];    // 24 KB (aliased for stats)
  __shared__ __align__(16) h16 sH[8 * 16 * M0];     // 16 KB
  int tid = (int)threadIdx.x;
  for (int i = tid; i < M0 * CP1; i += 256) {
    int r = i / CP1, k = i - r * CP1;
    sW0[i] = (h16)((k < CIN1) ? w1_0[r * CIN1 + k] : 0.f);
  }
  for (int i = tid; i < M1 * M0; i += 256) sW1[i] = (h16)w1_1[i];
  __syncthreads();

  int wave = tid >> 5, lane = tid & 31, hi = lane >> 4, col = lane & 15;
  int tile = (int)blockIdx.x * 8 + wave;
  int b = tile >> 13, n = tile & (N1 - 1);
  const float* f2b = feature2 + (size_t)b * 64 * N2;
  const float* p2b = pos2 + (size_t)b * 3 * N2;
  float p1x = pos1[(b * 3 + 0) * N1 + n];
  float p1y = pos1[(b * 3 + 1) * N1 + n];
  float p1z = pos1[(b * 3 + 2) * N1 + n];
  h16* xt = sX + wave * (16 * CP1);
  gather_x(xt, f2b, p2b, p1x, p1y, p1z, idx + (size_t)tile * KNNS);
  __syncthreads();

  v8f zero = {0.f, 0.f, 0.f, 0.f, 0.f, 0.f, 0.f, 0.f};
  v8f a0[4] = {zero, zero, zero, zero};
  mm_tiles<3, 4, CP1>(sW0, xt, a0);

  // BN0 fold + ReLU, stage h0 column-major (ld = 64) for layer-1 B fragments.
  h16* ht = sH + wave * (16 * M0);
#pragma unroll
  for (int mt = 0; mt < 4; ++mt) {
#pragma unroll
    for (int r = 0; r < 8; ++r) {
      int M = mt * 16 + hi * 8 + r;
      float y = fmaxf(fmaf(a0[mt][r], scale0[M], shift0[M]), 0.f);
      ht[col * M0 + M] = (h16)y;
    }
  }
  __syncthreads();   // sX dead past here; its LDS is reused for stats below.

  v8f a1[4] = {zero, zero, zero, zero};
  mm_tiles<2, 4, M0>(sW1, ht, a1);

  if (WRITE_F) {
#pragma unroll
    for (int mt = 0; mt < 4; ++mt) {
#pragma unroll
      for (int r = 0; r < 8; ++r) {
        int M = mt * 16 + hi * 8 + r;
        float y = fmaxf(fmaf(a1[mt][r], scale1[M], shift1[M]), 0.f);
#pragma unroll
        for (int o = 1; o < 16; o <<= 1) y = fmaxf(y, __shfl_xor(y, o, 32));
        if ((lane & 15) == 0) fbuf[(size_t)(b * 64 + M) * N1 + n] = y;
      }
    }
  } else {
    float* sS = (float*)sX;          // reuse gather LDS (safe: barrier above)
    float* sQ = sS + 8 * 64;
    wave_stats<4>(a1, sS + wave * 64, sQ + wave * 64);
    __syncthreads();
    if (tid < 64) {
      float s = 0.f, q = 0.f;
#pragma unroll
      for (int w = 0; w < 8; ++w) { s += sS[w * 64 + tid]; q += sQ[w * 64 + tid]; }
      psum[(size_t)blockIdx.x * 64 + tid] = s;
      psq [(size_t)blockIdx.x * 64 + tid] = q;
    }
  }
}

// --------------------------------------------------------------------- K5 ---
// y2 = W2 * [f; feature1]; write pre-BN y2 to d_out, emit per-block stats.
__global__ void __launch_bounds__(128)
k_l2(const float* __restrict__ fbuf, const float* __restrict__ feature1,
     const float* __restrict__ w2_0, float* __restrict__ out,
     float* __restrict__ psum, float* __restrict__ psq) {
  __shared__ __align__(16) h16 sW[M2 * CIN2];       // 32 KB
  __shared__ __align__(16) h16 sZ[4 * 16 * CIN2];   // 16 KB
  __shared__ float sS[4 * 128], sQ[4 * 128];        //  4 KB
  int tid = (int)threadIdx.x;
  for (int i = tid; i < M2 * CIN2; i += 128) sW[i] = (h16)w2_0[i];
  __syncthreads();

  int wave = tid >> 5, lane = tid & 31, hi = lane >> 4, c = lane & 15;
  int tile = (int)blockIdx.x * 4 + wave;
  int col0 = tile * 16;
  int b = col0 >> 13, n0 = col0 & (N1 - 1);
  h16* zt = sZ + wave * (16 * CIN2);
  for (int j = 0; j < 16; ++j) {
    int n = n0 + j;
#pragma unroll
    for (int k = lane; k < CIN2; k += 32) {
      float v = (k < 64) ? fbuf[(size_t)(b * 64 + k) * N1 + n]
                         : feature1[(size_t)(b * 64 + (k - 64)) * N1 + n];
      zt[j * CIN2 + k] = (h16)v;
    }
  }
  __syncthreads();

  v8f zero = {0.f, 0.f, 0.f, 0.f, 0.f, 0.f, 0.f, 0.f};
  v8f acc[8] = {zero, zero, zero, zero, zero, zero, zero, zero};
  mm_tiles<4, 8, CIN2>(sW, zt, acc);

#pragma unroll
  for (int mt = 0; mt < 8; ++mt) {
#pragma unroll
    for (int r = 0; r < 8; ++r) {
      int M = mt * 16 + hi * 8 + r;
      out[(size_t)(b * 128 + M) * N1 + n0 + c] = acc[mt][r];
    }
  }
  wave_stats<8>(acc, sS + wave * 128, sQ + wave * 128);
  __syncthreads();
  if (tid < 128) {
    float s = 0.f, q = 0.f;
#pragma unroll
    for (int w = 0; w < 4; ++w) { s += sS[w * 128 + tid]; q += sQ[w * 128 + tid]; }
    psum[(size_t)blockIdx.x * 128 + tid] = s;
    psq [(size_t)blockIdx.x * 128 + tid] = q;
  }
}

// ------------------------------------------------------------- K2b/3b/5b ---
__global__ void k_reduce_finalize(const float* __restrict__ psum,
                                  const float* __restrict__ psq, int nblk, int C,
                                  const float* __restrict__ g,
                                  const float* __restrict__ beta, float inv_cnt,
                                  float* __restrict__ scale,
                                  float* __restrict__ shift) {
  int c = (int)threadIdx.x;
  if (c >= C) return;
  float s = 0.f, q = 0.f;
  for (int i = 0; i < nblk; ++i) { s += psum[(size_t)i * C + c]; q += psq[(size_t)i * C + c]; }
  float m  = s * inv_cnt;
  float v  = q * inv_cnt - m * m;
  float rs = rsqrtf(v + BN_EPS);
  float sc = g[c] * rs;
  scale[c] = sc;
  shift[c] = beta[c] - m * sc;
}

// --------------------------------------------------------------------- K6 ---
__global__ void __launch_bounds__(256)
k_bnrelu(float* __restrict__ out, const float* __restrict__ scale,
         const float* __restrict__ shift) {
  int i = (int)blockIdx.x * 256 + (int)threadIdx.x;   // B*M2*N1 = 8,388,608
  int c = (i >> 13) & 127;
  out[i] = fmaxf(fmaf(out[i], scale[c], shift[c]), 0.f);
}

// ---------------------------------------------------------------------------
extern "C" void kernel_launch(void* const* d_in, const int* in_sizes, int n_in,
                              void* d_out, int out_size, void* d_ws, size_t ws_size,
                              hipStream_t stream) {
  (void)in_sizes; (void)n_in; (void)out_size; (void)ws_size;
  const float* pos1     = (const float*)d_in[0];
  const float* pos2     = (const float*)d_in[1];
  const float* feature1 = (const float*)d_in[2];
  const float* feature2 = (const float*)d_in[3];
  const float* w1_0 = (const float*)d_in[4];
  const float* g1_0 = (const float*)d_in[5];
  const float* b1_0 = (const float*)d_in[6];
  const float* w1_1 = (const float*)d_in[7];
  const float* g1_1 = (const float*)d_in[8];
  const float* b1_1 = (const float*)d_in[9];
  const float* w2_0 = (const float*)d_in[10];
  const float* g2_0 = (const float*)d_in[11];
  const float* b2_0 = (const float*)d_in[12];
  float* outf = (float*)d_out;

  // workspace layout (~25 MB total)
  char*  ws    = (char*)d_ws;
  int*   idxb  = (int*)ws;                                    // 4 MiB
  float* fbuf  = (float*)(ws + (size_t)B * N1 * KNNS * 4);    // 16 MiB
  float* psum  = fbuf + (size_t)B * 64 * N1;                  // 2 MiB (8192*64)
  float* psq   = psum + 8192 * 64;                            // 2 MiB
  float* par   = psq + 8192 * 64;                             // 512 floats
  float* scale0 = par + 0,   *shift0 = par + 64;
  float* scale1 = par + 128, *shift1 = par + 192;
  float* scale2 = par + 256, *shift2 = par + 384;

  const float invN01 = 1.f / (float)(B * N1 * KNNS);          // 1/1048576
  const float invN2  = 1.f / (float)(B * N1);                 // 1/65536

  k_knn<<<B * (N1 / 256), 256, 0, stream>>>(pos1, pos2, idxb);

  k_l0_stats<<<8192, 256, 0, stream>>>(pos1, pos2, feature2, w1_0, idxb,
                                       psum, psq);
  k_reduce_finalize<<<1, 64, 0, stream>>>(psum, psq, 8192, 64, g1_0, b1_0,
                                          invN01, scale0, shift0);

  k_mlp1<false><<<8192, 256, 0, stream>>>(pos1, pos2, feature2, w1_0, w1_1,
                                          idxb, scale0, shift0, nullptr,
                                          nullptr, psum, psq, nullptr);
  k_reduce_finalize<<<1, 64, 0, stream>>>(psum, psq, 8192, 64, g1_1, b1_1,
                                          invN01, scale1, shift1);

  k_mlp1<true><<<8192, 256, 0, stream>>>(pos1, pos2, feature2, w1_0, w1_1,
                                         idxb, scale0, shift0, scale1, shift1,
                                         nullptr, nullptr, fbuf);

  k_l2<<<1024, 128, 0, stream>>>(fbuf, feature1, w2_0, outf, psum, psq);
  k_reduce_finalize<<<1, 128, 0, stream>>>(psum, psq, 1024, 128, g2_0, b2_0,
                                           invN2, scale2, shift2);

  k_bnrelu<<<(B * M2 * N1) / 256, 256, 0, stream>>>(outf, scale2, shift2);
}
